// SelfAttention_58325655879829
// MI455X (gfx1250) — compile-verified, matrix operand order checked by reading, not run on hardware
//
#include <hip/hip_runtime.h>
#include <hip/hip_bf16.h>

// Self-attention: N=2, L=2048, EMBED=1024, H=16, D=64.
// (1) Q/K head projections -> bf16 (N,H,L,D); V projection -> bf16 (N,H,D,L)
//     (transposed so attention PV B-fragments are row-major),
// (2) flash attention with TDM (tensor_load_to_lds) staging of K / V^T tiles,
// (3) output projection + bias -> f32.
// GEMMs via v_wmma_f32_16x16x32_bf16 (f32 accumulate). All epilogues emit
// coalesced 128-bit stores (LDS-bounced where the C layout scatters).

#define L_SEQ 2048
#define NBATCH 2
#define HEADS 16
#define HDIM 64
#define EMBED 1024

typedef __attribute__((ext_vector_type(16))) __bf16 v16bf;
typedef __attribute__((ext_vector_type(8))) float v8f;
typedef __attribute__((ext_vector_type(4))) unsigned int u32x4;
typedef __attribute__((ext_vector_type(8))) int i32x8;
typedef __attribute__((ext_vector_type(4))) int i32x4;

#if defined(__has_builtin)
#if __has_builtin(__builtin_amdgcn_tensor_load_to_lds) && \
    __has_builtin(__builtin_amdgcn_s_wait_tensorcnt)
#define HAVE_TDM 1
#endif
#endif

union FragU {
  v16bf v;
  uint4 q[2];
};
union PackU {
  uint4 q;
  __bf16 h[8];
};

// Fragment element map (ISA 7.12.2, 16-bit A 16x32): lane = (half<<4)|m,
// element e holds K = (e&8)*2 + half*8 + (e&7): elements 0..7 = 8 contiguous
// bf16 at K = half*8, elements 8..15 = 8 contiguous at K = 16 + half*8.

// Row-major bf16 source: two 16-byte vector loads per lane (global or LDS).
__device__ inline v16bf load_fragA_bf16_rm(const __bf16* __restrict__ base,
                                           int rs, int lane) {
  const int m = lane & 15;
  const int kh = (lane >> 4) << 3;
  const __bf16* p = base + (size_t)m * rs + kh;
  FragU f;
  f.q[0] = *(const uint4*)p;
  f.q[1] = *(const uint4*)(p + 16);
  return f.v;
}

// Row-major f32 source: contiguous f32 loads (vectorized) + pack-convert.
__device__ inline v16bf load_fragA_f32_rm(const float* __restrict__ base,
                                          int rs, int lane) {
  const int m = lane & 15;
  const int kh = (lane >> 4) << 3;
  const float* p = base + (size_t)m * rs + kh;
  v16bf a;
#pragma unroll
  for (int e = 0; e < 8; ++e) {
    a[e] = (__bf16)p[e];
    a[e + 8] = (__bf16)p[e + 16];
  }
  return a;
}

__device__ inline v8f wmma_bf16(v16bf a, v16bf b, v8f c) {
  return __builtin_amdgcn_wmma_f32_16x16x32_bf16(false, a, false, b, (short)0, c,
                                                 false, false);
}

__device__ inline float hmax16(float v) {
#pragma unroll
  for (int m = 8; m >= 1; m >>= 1) v = fmaxf(v, __shfl_xor(v, m, 16));
  return v;
}
__device__ inline float hsum16(float v) {
#pragma unroll
  for (int m = 8; m >= 1; m >>= 1) v += __shfl_xor(v, m, 16);
  return v;
}

#ifdef HAVE_TDM
// Issue one TDM 2D tile load: tile_d1 rows of tile_d0 bf16 elements, row
// stride stride_d0 (elements), packed row-major into LDS at lds_off.
// D# layout per CDNA5 ISA §8.3/8.4. 6-arg builtin on this toolchain:
//   (u32x4 g0, i32x8 g1, i32x4 g2, i32x4 g3, i32x8 gx, i32 cpol).
__device__ inline void tdm_load_2d(unsigned lds_off, const __bf16* gptr,
                                   unsigned tensor_d0, unsigned tensor_d1,
                                   unsigned tile_d0, unsigned tile_d1,
                                   unsigned stride_d0) {
  const unsigned long long ga = (unsigned long long)(uintptr_t)gptr;
  u32x4 g0;
  g0[0] = 1u;                                  // count=1 (valid user D#)
  g0[1] = lds_off;                             // lds_addr (bytes)
  g0[2] = (unsigned)ga;                        // global_addr[31:0]
  g0[3] = (unsigned)(ga >> 32) | (2u << 30);   // global_addr[56:32] | type=2
  i32x8 g1;
  g1[0] = 1 << 16;                             // data_size = 1 -> 2 bytes
  g1[1] = (int)(tensor_d0 << 16);              // tensor_dim0[15:0] -> [31:16]
  g1[2] = (int)((tensor_d0 >> 16) | (tensor_d1 << 16));
  g1[3] = (int)((tensor_d1 >> 16) | (tile_d0 << 16));
  g1[4] = (int)(tile_d1 & 0xFFFFu);            // tile_dim1 (tile_dim2 = 0)
  g1[5] = (int)stride_d0;                      // tensor_dim0_stride[31:0]
  g1[6] = 0;
  g1[7] = 0;
  const i32x4 gz4 = {0, 0, 0, 0};
  const i32x8 gz8 = {0, 0, 0, 0, 0, 0, 0, 0};
  __builtin_amdgcn_tensor_load_to_lds(g0, g1, gz4, gz4, gz8, 0);
}
#endif

// ---------------- Kernel 1a: Q/K head projection ----------------------------
// Wave tile: one token x 16 heads (rows) x 64 dims. C tile bounced through
// LDS, then 4 coalesced global_store_b128 per lane into (N,H,L,D).
// grid = (512, 2: q/k), block = 256 (8 waves).
__global__ void proj_qk_kernel(const float* __restrict__ keys,
                               const float* __restrict__ qrys,
                               const float* __restrict__ Wk,
                               const float* __restrict__ Wq,
                               unsigned short* __restrict__ kp_u,
                               unsigned short* __restrict__ qp_u) {
  __shared__ __align__(16) __bf16 cbuf[8][16 * HDIM];
  const int lane = threadIdx.x & 31;
  const int w = threadIdx.x >> 5;
  const float* x;
  const float* W;
  __bf16* dst;
  if (blockIdx.y == 0) { x = qrys; W = Wq; dst = (__bf16*)qp_u; }
  else { x = keys; W = Wk; dst = (__bf16*)kp_u; }

  const int r0 = (blockIdx.x * 8 + w) * 16;  // rows r = token*16 + h
  const int token = r0 >> 4;
  const int n = token / L_SEQ, l = token - n * L_SEQ;
  const int half8 = (lane >> 4) << 3;
  const int coll = lane & 15;

  v16bf a0 = load_fragA_f32_rm(x + (size_t)r0 * HDIM, HDIM, lane);
  v16bf a1 = load_fragA_f32_rm(x + (size_t)r0 * HDIM + 32, HDIM, lane);

#pragma unroll
  for (int t = 0; t < 4; ++t) {
    v16bf b0 = load_fragA_f32_rm(W + (size_t)t * 16 * HDIM, HDIM, lane);
    v16bf b1 = load_fragA_f32_rm(W + (size_t)t * 16 * HDIM + 32, HDIM, lane);
    v8f c = {};
    c = wmma_bf16(a0, b0, c);
    c = wmma_bf16(a1, b1, c);
#pragma unroll
    for (int i = 0; i < 8; ++i)
      cbuf[w][(half8 + i) * HDIM + t * 16 + coll] = (__bf16)c[i];
  }
  // Coalesced stores: tile = 128 uint4; each lane stores 4 (row h contiguous).
  const uint4* sb = (const uint4*)cbuf[w];
#pragma unroll
  for (int qq = 0; qq < 4; ++qq) {
    const int f = lane * 4 + qq;
    const int row = f >> 3, seg = f & 7;  // row = h, seg = 8-elem chunk of d
    const size_t idx = (((size_t)n * HEADS + row) * L_SEQ + l) * HDIM + seg * 8;
    *(uint4*)(dst + idx) = sb[f];
  }
}

// ---------------- Kernel 1b: V head projection (transposed out) -------------
// Wave tile: 16 consecutive tokens x one head x 64 dims. Each lane's 8 C
// values are consecutive-l for one output column -> single b128 store into
// V^T (N,H,D,L). grid = 512, block = 256 (8 waves).
__global__ void proj_v_kernel(const float* __restrict__ vals,
                              const float* __restrict__ Wv,
                              unsigned short* __restrict__ vp_u) {
  const int lane = threadIdx.x & 31;
  const int w = threadIdx.x >> 5;
  const int wg = blockIdx.x * 8 + w;   // 4096 wave tiles
  const int h = wg & 15;
  const int tk0 = (wg >> 4) * 16;      // 16 consecutive tokens
  const int n = tk0 / L_SEQ;
  const int l0 = tk0 - n * L_SEQ;
  const int half8 = (lane >> 4) << 3;
  __bf16* Vt = (__bf16*)vp_u;

  const float* xb = vals + (size_t)tk0 * EMBED + h * HDIM;  // row stride EMBED
  v16bf a0 = load_fragA_f32_rm(xb, EMBED, lane);
  v16bf a1 = load_fragA_f32_rm(xb + 32, EMBED, lane);

#pragma unroll
  for (int t = 0; t < 4; ++t) {
    v16bf b0 = load_fragA_f32_rm(Wv + (size_t)t * 16 * HDIM, HDIM, lane);
    v16bf b1 = load_fragA_f32_rm(Wv + (size_t)t * 16 * HDIM + 32, HDIM, lane);
    v8f c = {};
    c = wmma_bf16(a0, b0, c);
    c = wmma_bf16(a1, b1, c);
    PackU pk;
#pragma unroll
    for (int i = 0; i < 8; ++i) pk.h[i] = (__bf16)c[i];  // l = l0+half8+i
    const int col = t * 16 + (lane & 15);                // output dim d
    const size_t idx =
        ((size_t)(n * HEADS + h) * HDIM + col) * L_SEQ + l0 + half8;
    *(uint4*)(Vt + idx) = pk.q;
  }
}

// ---------------- Kernel 2: flash attention ---------------------------------
// grid.x = N*H = 32, grid.y = L/64 = 32, block = 128 (4 waves, 16 q-rows each).
// K tile (32x64) and V^T tile (64x32) staged in LDS via TDM, shared by waves.
__global__ void attn_kernel(const unsigned short* __restrict__ qp_u,
                            const unsigned short* __restrict__ kp_u,
                            const unsigned short* __restrict__ vp_u,
                            unsigned short* __restrict__ obuf_u) {
  const int tid = threadIdx.x;
  const int lane = tid & 31;
  const int w = tid >> 5;  // 0..3
  const int nh = blockIdx.x;
  const int n = nh >> 4, h = nh & 15;
  const int q0 = blockIdx.y * 64 + w * 16;
  const int half8 = (lane >> 4) << 3;
  const int coll = lane & 15;

  const __bf16* Q = (const __bf16*)qp_u + (size_t)nh * L_SEQ * HDIM;
  const __bf16* K = (const __bf16*)kp_u + (size_t)nh * L_SEQ * HDIM;
  const __bf16* Vt = (const __bf16*)vp_u + (size_t)nh * HDIM * L_SEQ;  // (D,L)
  __bf16* obuf = (__bf16*)obuf_u;

  __shared__ __align__(16) __bf16 kt[32 * HDIM];     // K tile: [j][d]
  __shared__ __align__(16) __bf16 vtt[HDIM * 32];    // V^T tile: [d][j]
  __shared__ __align__(16) __bf16 pbuf[4][16 * 32];  // per-wave P staging

  const v16bf qa0 = load_fragA_bf16_rm(Q + (size_t)q0 * HDIM, HDIM, lane);
  const v16bf qa1 = load_fragA_bf16_rm(Q + (size_t)q0 * HDIM + 32, HDIM, lane);

  const v8f vzero = {};
  v8f acc[4];
#pragma unroll
  for (int t = 0; t < 4; ++t) acc[t] = vzero;
  float rm[8], rl[8];
#pragma unroll
  for (int i = 0; i < 8; ++i) { rm[i] = -1e30f; rl[i] = 0.0f; }

  const float sc = 0.03125f;  // 1/sqrt(EMBED) = 1/32

  for (int j0 = 0; j0 < L_SEQ; j0 += 32) {
#ifdef HAVE_TDM
    if (w == 0) {
      // K tile: 32 rows x 64 elems, row stride 64 (contiguous block).
      tdm_load_2d((unsigned)(uintptr_t)&kt[0], K + (size_t)j0 * HDIM,
                  HDIM, L_SEQ, HDIM, 32, HDIM);
      // V^T tile: 64 rows (d) x 32 elems (j), row stride L_SEQ.
      tdm_load_2d((unsigned)(uintptr_t)&vtt[0], Vt + j0,
                  L_SEQ, HDIM, 32, HDIM, L_SEQ);
      __builtin_amdgcn_s_wait_tensorcnt((short)0);
    }
#else
    {  // Cooperative staging: 256 uint4 per tile, 128 threads.
      const uint4* gk = (const uint4*)(K + (size_t)j0 * HDIM);
      uint4* sk = (uint4*)kt;
      sk[tid * 2] = gk[tid * 2];
      sk[tid * 2 + 1] = gk[tid * 2 + 1];
      uint4* sv = (uint4*)vtt;
#pragma unroll
      for (int c = tid * 2; c < tid * 2 + 2; ++c) {
        const int row = c >> 2, seg = c & 3;
        sv[c] = *(const uint4*)(Vt + (size_t)row * L_SEQ + j0 + seg * 8);
      }
    }
#endif
    // Prefetch next tiles into cache (global_prefetch_b8).
    if (j0 + 32 < L_SEQ) {
      __builtin_prefetch(K + (size_t)(j0 + 32) * HDIM + tid * 16, 0, 3);
      __builtin_prefetch(Vt + (size_t)(tid >> 1) * L_SEQ + j0 + 32 +
                             (tid & 1) * 16, 0, 3);
    }
    __syncthreads();

    // S = Q * K^T: 16 q-rows x 32 k-cols (two 16x16 C tiles); B^T = K rows.
    v8f s[2];
#pragma unroll
    for (int t = 0; t < 2; ++t) {
      v16bf b0 = load_fragA_bf16_rm(kt + (size_t)t * 16 * HDIM, HDIM, lane);
      v16bf b1 = load_fragA_bf16_rm(kt + (size_t)t * 16 * HDIM + 32, HDIM, lane);
      v8f c = {};
      c = wmma_bf16(qa0, b0, c);
      c = wmma_bf16(qa1, b1, c);
      s[t] = c;
    }
    // Online softmax per row (row = half8 + i, replicated across lane-half).
#pragma unroll
    for (int i = 0; i < 8; ++i) {
      const float v0 = s[0][i] * sc;
      const float v1 = s[1][i] * sc;
      const float tmax = hmax16(fmaxf(v0, v1));
      const float mnew = fmaxf(rm[i], tmax);
      const float scale = __expf(rm[i] - mnew);
      rm[i] = mnew;
      const float p0 = __expf(v0 - mnew);
      const float p1 = __expf(v1 - mnew);
      rl[i] = rl[i] * scale + hsum16(p0 + p1);
#pragma unroll
      for (int t = 0; t < 4; ++t) acc[t][i] *= scale;
      pbuf[w][(half8 + i) * 32 + coll] = (__bf16)p0;
      pbuf[w][(half8 + i) * 32 + 16 + coll] = (__bf16)p1;
    }
    // O += P * V. P as A fragment; B^T = V^T rows from vtt (all ds_load_b128).
    const v16bf pa = load_fragA_bf16_rm(pbuf[w], 32, lane);
#pragma unroll
    for (int t = 0; t < 4; ++t) {
      v16bf vb = load_fragA_bf16_rm(vtt + (size_t)t * 16 * 32, 32, lane);
      acc[t] = wmma_bf16(pa, vb, acc[t]);
    }
    __syncthreads();  // protect kt/vtt before next iteration's staging
  }

  // Epilogue: normalize, bounce 16x64 tile through LDS (kt/vtt are free after
  // the final barrier), then 4 coalesced b128 stores per lane into obuf.
  __bf16* ob = (w < 2 ? kt : vtt) + (size_t)(w & 1) * (16 * HDIM);
#pragma unroll
  for (int i = 0; i < 8; ++i) {
    const float inv = 1.0f / rl[i];
#pragma unroll
    for (int t = 0; t < 4; ++t)
      ob[(half8 + i) * HDIM + t * 16 + coll] = (__bf16)(acc[t][i] * inv);
  }
  const uint4* sb = (const uint4*)ob;
#pragma unroll
  for (int qq = 0; qq < 4; ++qq) {
    const int f = lane * 4 + qq;
    const int row = f >> 3, seg = f & 7;
    const int l = q0 + row;
    const size_t idx = ((size_t)n * L_SEQ + l) * EMBED + h * HDIM + seg * 8;
    *(uint4*)(obuf + idx) = sb[f];
  }
}

// ---------------- Kernel 3: output projection -------------------------------
// out = O (4096x1024, bf16) @ Wo^T + bo, f32 out. B^T = Wo rows (f32->bf16).
// grid = (4096/32, 1024/64), block = 256 (8 waves: 2 row-tiles x 4 col-tiles).
__global__ void outproj_kernel(const unsigned short* __restrict__ obuf_u,
                               const float* __restrict__ Wo,
                               const float* __restrict__ bo,
                               float* __restrict__ out) {
  __shared__ __align__(16) float fb[8][16 * 16];
  const int lane = threadIdx.x & 31;
  const int w = threadIdx.x >> 5;
  const int wr = w >> 2, wc = w & 3;
  const int row0 = blockIdx.x * 32 + wr * 16;
  const int col0 = blockIdx.y * 64 + wc * 16;
  const int half8 = (lane >> 4) << 3;
  const int coll = lane & 15;
  const __bf16* O = (const __bf16*)obuf_u;

  v8f c = {};
  for (int k0 = 0; k0 < EMBED; k0 += 32) {
    v16bf a = load_fragA_bf16_rm(O + (size_t)row0 * EMBED + k0, EMBED, lane);
    v16bf b = load_fragA_f32_rm(Wo + (size_t)col0 * EMBED + k0, EMBED, lane);
    c = wmma_bf16(a, b, c);
  }
  const float bias = bo[col0 + coll];
#pragma unroll
  for (int i = 0; i < 8; ++i) fb[w][(half8 + i) * 16 + coll] = c[i] + bias;
  // Coalesced f32 stores: tile = 64 uint4; each lane stores 2.
  const uint4* sb = (const uint4*)fb[w];
#pragma unroll
  for (int qq = 0; qq < 2; ++qq) {
    const int f = lane * 2 + qq;
    const int row = f >> 2, seg = f & 3;
    const size_t idx = (size_t)(row0 + row) * EMBED + col0 + seg * 4;
    *(uint4*)(out + idx) = sb[f];
  }
}

extern "C" void kernel_launch(void* const* d_in, const int* in_sizes, int n_in,
                              void* d_out, int out_size, void* d_ws, size_t ws_size,
                              hipStream_t stream) {
  const float* values  = (const float*)d_in[0];
  const float* keys    = (const float*)d_in[1];
  const float* queries = (const float*)d_in[2];
  const float* Wv = (const float*)d_in[3];
  const float* Wk = (const float*)d_in[4];
  const float* Wq = (const float*)d_in[5];
  const float* Wo = (const float*)d_in[6];
  const float* bo = (const float*)d_in[7];
  float* out = (float*)d_out;

  // Workspace (bf16 elements): qp | kp | vp | obuf, each N*L*EMBED = 4M elems.
  unsigned short* ws = (unsigned short*)d_ws;
  const size_t TEN = (size_t)NBATCH * L_SEQ * EMBED;  // 4,194,304
  unsigned short* qp = ws;
  unsigned short* kp = ws + TEN;
  unsigned short* vp = ws + 2 * TEN;
  unsigned short* obuf = ws + 3 * TEN;

  proj_qk_kernel<<<dim3(512, 2), 256, 0, stream>>>(keys, queries, Wk, Wq, kp, qp);
  proj_v_kernel<<<512, 256, 0, stream>>>(values, Wv, vp);
  attn_kernel<<<dim3(NBATCH * HEADS, L_SEQ / 64), 128, 0, stream>>>(qp, kp, vp, obuf);
  outproj_kernel<<<dim3((NBATCH * L_SEQ) / 32, EMBED / 64), 256, 0, stream>>>(
      obuf, Wo, bo, out);
}
